// SelectiveSSM_53068615909664
// MI455X (gfx1250) — matrix operand order repaired and checked by reading, not compiled, not run on hardware
//
#include <hip/hip_runtime.h>

// ---------------------------------------------------------------------------
// Shapes (fixed by the reference)
// ---------------------------------------------------------------------------
#define BB   4
#define LL   2048
#define DM   1024
#define DS   16
#define DC   4
#define DI   2048            // DM * EX
#define MROWS (BB * LL)      // 8192

typedef __attribute__((ext_vector_type(16))) _Float16 v16h;
typedef __attribute__((ext_vector_type(8)))  _Float16 v8h;
typedef __attribute__((ext_vector_type(4)))  _Float16 v4h;
typedef __attribute__((ext_vector_type(8)))  float    v8f;

// LDS byte offset of a __shared__ object (generic -> addrspace(3) -> int)
__device__ __forceinline__ unsigned lds_off(const void* p) {
    return (unsigned)(size_t)(const __attribute__((address_space(3))) void*)p;
}

// One 16-byte global -> LDS async copy per lane (ASYNCcnt-tracked)
__device__ __forceinline__ void async_ld_b128(unsigned dst_lds, const void* src) {
    asm volatile("global_load_async_to_lds_b128 %0, %1, off"
                 :: "v"(dst_lds), "v"(src) : "memory");
}
__device__ __forceinline__ void wait_async0() {
    asm volatile("s_wait_asynccnt 0x0" ::: "memory");
}

#define WMMA_F16(a, b, c) __builtin_amdgcn_wmma_f32_16x16x32_f16( \
        false, (a), false, (b), (short)0, (c), false, false)

// ---------------------------------------------------------------------------
// WMMA GEMM:  C[M,N] = A[M,K] * W[N,K]^T   (A, W pre-converted to f16 and
// K-contiguous; f32 accumulate).
//
// Block tile 128(M) x 64(N), K-tile 64 (2 WMMA K-steps). 256 threads =
// 8 waves laid out 4(M) x 2(N); each wave owns a 32x32 output tile
// (4 accumulators) -> 8 x v_wmma_f32_16x16x32_f16 per K-tile, and
// 4 fragment loads feed 4 WMMAs per K-step (A fragments reused across N,
// B fragments across M).
//
// LDS tiles live in the ISA 16-bit WMMA fragment layout, so each 8-half
// segment is 16 contiguous bytes in both global memory and LDS: staged with
// one global_load_async_to_lds_b128 per lane-segment, double-buffered with
// fixed buffer indices (2x-unrolled pipeline: issue tile t+1, compute
// tile t, s_wait_asynccnt 0, barrier). K/64 is even for both GEMMs.
// Fragment reads are two 16B ds_load_b128 per operand (48B lane stride ->
// lanes 0-15 tile all 64 banks, lanes 16-31 are pass 2).
// ---------------------------------------------------------------------------
__global__ __launch_bounds__(256) void ssm_gemm_f16_tn(
    const _Float16* __restrict__ A, const _Float16* __restrict__ W,
    float* __restrict__ C, int M, int N, int K)
{
    // [buf][kstep][16-wide tile][lane][24 halves: 16 used, pad to 48B stride]
    __shared__ __align__(16) _Float16 Af[2][2][8][32][24];   // 128 M-rows
    __shared__ __align__(16) _Float16 Bf[2][2][4][32][24];   // 64  N-rows

    const int tid  = threadIdx.x;
    const int wave = tid >> 5;
    const int lane = tid & 31;

    const int m0 = blockIdx.y * 128;
    const int n0 = blockIdx.x * 64;

    const int mt0 = (wave >> 1) * 2;      // wave's first M 16-tile (0,2,4,6)
    const int mt1 = mt0 + 1;
    const int ni  = wave & 1;             // wave's N 32-strip
    const int nt0 = ni * 2;
    const int nt1 = nt0 + 1;
    const int ncol0 = n0 + ni * 32;

    // ---- A staging: 2 threads per row (128 rows), 32 consecutive K halves
    const int rowa = tid >> 1;            // 0..127
    const int kca  = (tid & 1) * 32;      // 0,32
    const int ksa  = kca >> 5;            // K-step
    const int rta  = rowa >> 4;           // 16-wide tile 0..7
    const int rla  = rowa & 15;
    // ---- B staging: 4 threads per row (64 rows), 16 consecutive K halves
    const int rowb = tid >> 2;            // 0..63
    const int kcb  = (tid & 3) * 16;      // 0,16,32,48
    const int ksb  = kcb >> 5;
    const int rtb  = rowb >> 4;
    const int rlb  = rowb & 15;

    // per-thread LDS destinations (fragment-layout scatter), per buffer
    unsigned dA[2][4], dB[2][2];
    #pragma unroll
    for (int p = 0; p < 2; ++p) {
        dA[p][0] = lds_off(&Af[p][ksa][rta][rla     ][0]);  // k' 0..7
        dA[p][1] = lds_off(&Af[p][ksa][rta][rla + 16][0]);  // k' 8..15
        dA[p][2] = lds_off(&Af[p][ksa][rta][rla     ][8]);  // k' 16..23
        dA[p][3] = lds_off(&Af[p][ksa][rta][rla + 16][8]);  // k' 24..31
        dB[p][0] = lds_off(&Bf[p][ksb][rtb][rlb     ][(kcb & 16) >> 1]);
        dB[p][1] = lds_off(&Bf[p][ksb][rtb][rlb + 16][(kcb & 16) >> 1]);
    }
    const _Float16* gA = &A[(size_t)(m0 + rowa) * K + kca];
    const _Float16* gW = &W[(size_t)(n0 + rowb) * K + kcb];

    v8f acc00 = {}, acc01 = {}, acc10 = {}, acc11 = {};

    auto stage = [&](int p, int k0) {
        const _Float16* sA = gA + k0;
        async_ld_b128(dA[p][0], sA);
        async_ld_b128(dA[p][1], sA + 8);
        async_ld_b128(dA[p][2], sA + 16);
        async_ld_b128(dA[p][3], sA + 24);
        const _Float16* sW = gW + k0;
        async_ld_b128(dB[p][0], sW);
        async_ld_b128(dB[p][1], sW + 8);
    };
    auto compute = [&](int p) {
        #pragma unroll
        for (int s = 0; s < 2; ++s) {
            union { v16h v; v8h h[2]; } a0, a1, b0, b1;
            a0.h[0] = *reinterpret_cast<const v8h*>(&Af[p][s][mt0][lane][0]);
            a0.h[1] = *reinterpret_cast<const v8h*>(&Af[p][s][mt0][lane][8]);
            a1.h[0] = *reinterpret_cast<const v8h*>(&Af[p][s][mt1][lane][0]);
            a1.h[1] = *reinterpret_cast<const v8h*>(&Af[p][s][mt1][lane][8]);
            b0.h[0] = *reinterpret_cast<const v8h*>(&Bf[p][s][nt0][lane][0]);
            b0.h[1] = *reinterpret_cast<const v8h*>(&Bf[p][s][nt0][lane][8]);
            b1.h[0] = *reinterpret_cast<const v8h*>(&Bf[p][s][nt1][lane][0]);
            b1.h[1] = *reinterpret_cast<const v8h*>(&Bf[p][s][nt1][lane][8]);
            acc00 = WMMA_F16(a0.v, b0.v, acc00);
            acc01 = WMMA_F16(a0.v, b1.v, acc01);
            acc10 = WMMA_F16(a1.v, b0.v, acc10);
            acc11 = WMMA_F16(a1.v, b1.v, acc11);
        }
    };

    // ---- software pipeline, fixed buffer indices, no branches in the loop
    const int ntiles = K >> 6;            // even (16 or 32)
    stage(0, 0);
    wait_async0();
    __syncthreads();

    for (int t = 0; t + 2 < ntiles; t += 2) {
        stage(1, (t + 1) << 6);
        compute(0);
        wait_async0();
        __syncthreads();
        stage(0, (t + 2) << 6);
        compute(1);
        wait_async0();
        __syncthreads();
    }
    stage(1, (ntiles - 1) << 6);
    compute(0);
    wait_async0();
    __syncthreads();
    compute(1);

    // ---- store C per 16x16 f32 C/D layout (VGPR r: M=r / M=8+r) ----
    const int col   = lane & 15;
    const int rbase = (lane >= 16) ? 8 : 0;
    const int mrow0 = m0 + mt0 * 16;
    const int mrow1 = m0 + mt1 * 16;
    #pragma unroll
    for (int r = 0; r < 8; ++r) {
        C[(size_t)(mrow0 + rbase + r) * N + ncol0 + col]      = acc00[r];
        C[(size_t)(mrow0 + rbase + r) * N + ncol0 + 16 + col] = acc01[r];
        C[(size_t)(mrow1 + rbase + r) * N + ncol0 + col]      = acc10[r];
        C[(size_t)(mrow1 + rbase + r) * N + ncol0 + 16 + col] = acc11[r];
    }
}

// ---------------------------------------------------------------------------
// fp32 -> f16 conversion, 4 elements per thread (n multiple of 4)
// ---------------------------------------------------------------------------
__global__ void ssm_cvt_f16(const float* __restrict__ src,
                            _Float16* __restrict__ dst, size_t n4)
{
    size_t i = (size_t)blockIdx.x * blockDim.x + threadIdx.x;
    if (i >= n4) return;
    const float4 f = reinterpret_cast<const float4*>(src)[i];
    v4h h;
    h[0] = (_Float16)f.x; h[1] = (_Float16)f.y;
    h[2] = (_Float16)f.z; h[3] = (_Float16)f.w;
    reinterpret_cast<v4h*>(dst)[i] = h;
}

// ---------------------------------------------------------------------------
// Depthwise causal conv (DC=4 taps) + SiLU over the x_ssm half of x_inner.
// ---------------------------------------------------------------------------
__global__ void ssm_conv_silu(const float* __restrict__ x_inner,
                              const float* __restrict__ cw,
                              const float* __restrict__ cb,
                              float* __restrict__ x_conv)
{
    size_t i = (size_t)blockIdx.x * blockDim.x + threadIdx.x;
    if (i >= (size_t)MROWS * DI) return;
    const int d = (int)(i % DI);
    const size_t m = i / DI;            // m = b*L + l
    const int l = (int)(m % LL);
    float acc = cb[d];
    #pragma unroll
    for (int k = 0; k < DC; ++k) {
        const int ls = l - (DC - 1) + k;
        if (ls >= 0)
            acc += x_inner[(m + (size_t)(ls - l)) * (2 * DI) + d] * cw[d * DC + k];
    }
    x_conv[i] = acc / (1.0f + __expf(-acc));      // silu
}

// ---------------------------------------------------------------------------
// Fused delta / B / C projections: out[m][0:16)=delta_raw, [16:32)=Bs, [32:48)=Cs
// ---------------------------------------------------------------------------
__global__ __launch_bounds__(64) void ssm_proj_dbc(
    const float* __restrict__ xc,
    const float* __restrict__ dw, const float* __restrict__ db,
    const float* __restrict__ bw, const float* __restrict__ cw,
    float* __restrict__ dbc)
{
    const int m = blockIdx.x;
    const int j = threadIdx.x;
    if (j >= 48) return;
    const float* w;
    float bias = 0.0f;
    if (j < 16)      { w = &dw[(size_t)j * DI];        bias = db[j]; }
    else if (j < 32) { w = &bw[(size_t)(j - 16) * DI]; }
    else             { w = &cw[(size_t)(j - 32) * DI]; }
    const float* xr = &xc[(size_t)m * DI];
    float s = bias;
    const float4* x4 = reinterpret_cast<const float4*>(xr);
    const float4* w4 = reinterpret_cast<const float4*>(w);
    for (int k = 0; k < DI / 4; ++k) {
        float4 a = x4[k], b = w4[k];
        s += a.x * b.x + a.y * b.y + a.z * b.z + a.w * b.w;
    }
    dbc[(size_t)m * 48 + j] = s;
}

// ---------------------------------------------------------------------------
// Row mean of x_conv -> xavg[m]
// ---------------------------------------------------------------------------
__global__ __launch_bounds__(256) void ssm_row_mean(const float* __restrict__ xc,
                                                    float* __restrict__ xavg)
{
    __shared__ float red[256];
    const int m = blockIdx.x;
    float s = 0.0f;
    for (int k = threadIdx.x; k < DI; k += 256)
        s += xc[(size_t)m * DI + k];
    red[threadIdx.x] = s;
    __syncthreads();
    for (int off = 128; off > 0; off >>= 1) {
        if (threadIdx.x < (unsigned)off) red[threadIdx.x] += red[threadIdx.x + off];
        __syncthreads();
    }
    if (threadIdx.x == 0) xavg[m] = red[0] * (1.0f / DI);
}

// ---------------------------------------------------------------------------
// Sequential selective scan: one wave32 per batch, lanes 0..15 carry state s.
// ---------------------------------------------------------------------------
__global__ __launch_bounds__(32) void ssm_scan(const float* __restrict__ dbc,
                                               const float* __restrict__ xavg,
                                               const float* __restrict__ A_log,
                                               float* __restrict__ y)
{
    const int b = blockIdx.x;
    const int s = threadIdx.x;           // wave32; lanes 16..31 are passengers
    const float A = (s < DS) ? -__expf(A_log[s]) : 0.0f;
    float h = 0.0f;
    for (int t = 0; t < LL; ++t) {
        const size_t m = (size_t)b * LL + t;
        float dt = 0.0f, bt = 0.0f, ct = 0.0f;
        if (s < DS) {
            const float* row = &dbc[m * 48];
            const float raw = row[s];
            dt = (raw > 20.0f) ? raw : log1pf(__expf(raw));   // softplus
            bt = row[16 + s];
            ct = row[32 + s];
        }
        const float xa = xavg[m];
        h = __expf(dt * A) * h + dt * bt * xa;
        float p = h * ct;
        p += __shfl_xor(p, 1, 32);
        p += __shfl_xor(p, 2, 32);
        p += __shfl_xor(p, 4, 32);
        p += __shfl_xor(p, 8, 32);
        if (s == 0) y[m] = p;
    }
}

// ---------------------------------------------------------------------------
// y_skip = y_ssm * silu(x_gate) + x_conv * D   (emitted as f16 for out_proj)
// ---------------------------------------------------------------------------
__global__ void ssm_gate_skip(const float* __restrict__ x_inner,
                              const float* __restrict__ x_conv,
                              const float* __restrict__ y,
                              const float* __restrict__ Dp,
                              _Float16* __restrict__ y_skip)
{
    size_t i = (size_t)blockIdx.x * blockDim.x + threadIdx.x;
    if (i >= (size_t)MROWS * DI) return;
    const int d = (int)(i % DI);
    const size_t m = i / DI;
    const float g  = x_inner[m * (2 * DI) + DI + d];
    const float sg = g / (1.0f + __expf(-g));
    y_skip[i] = (_Float16)(y[m] * sg + x_conv[i] * Dp[d]);
}

// ---------------------------------------------------------------------------
// Launch
// ---------------------------------------------------------------------------
extern "C" void kernel_launch(void* const* d_in, const int* in_sizes, int n_in,
                              void* d_out, int out_size, void* d_ws, size_t ws_size,
                              hipStream_t stream)
{
    const float* x        = (const float*)d_in[0];   // (B,L,DM)
    const float* in_w     = (const float*)d_in[1];   // (2*DI, DM)
    const float* conv_w   = (const float*)d_in[2];   // (DI, DC)
    const float* conv_b   = (const float*)d_in[3];   // (DI,)
    const float* A_log    = (const float*)d_in[4];   // (DS,)
    const float* Dp       = (const float*)d_in[5];   // (DI,)
    const float* delta_w  = (const float*)d_in[6];   // (DS, DI)
    const float* delta_b  = (const float*)d_in[7];   // (DS,)
    const float* B_w      = (const float*)d_in[8];   // (DS, DI)
    const float* C_w      = (const float*)d_in[9];   // (DS, DI)
    const float* out_w    = (const float*)d_in[10];  // (DM, DI)
    float* out = (float*)d_out;                      // (B,L,DM)

    // workspace carve-up
    float* ws      = (float*)d_ws;
    float* x_inner = ws;                                   // M x 2*DI  (fp32)
    float* x_conv  = x_inner + (size_t)MROWS * 2 * DI;     // M x DI    (fp32)
    float* dbc     = x_conv  + (size_t)MROWS * DI;         // M x 48
    float* xavg    = dbc     + (size_t)MROWS * 48;         // M
    float* yssm    = xavg    + (size_t)MROWS;              // M
    _Float16* x16     = (_Float16*)(yssm + MROWS);         // M x DM
    _Float16* inw16   = x16    + (size_t)MROWS * DM;       // 2*DI x DM
    _Float16* outw16  = inw16  + (size_t)(2 * DI) * DM;    // DM x DI
    _Float16* yskip16 = outw16 + (size_t)DM * DI;          // M x DI

    // 0) one-time fp32 -> f16 down-conversion of GEMM operands
    {
        size_t n4;
        n4 = (size_t)MROWS * DM / 4;
        ssm_cvt_f16<<<(unsigned)((n4 + 255) / 256), 256, 0, stream>>>(x, x16, n4);
        n4 = (size_t)(2 * DI) * DM / 4;
        ssm_cvt_f16<<<(unsigned)((n4 + 255) / 256), 256, 0, stream>>>(in_w, inw16, n4);
        n4 = (size_t)DM * DI / 4;
        ssm_cvt_f16<<<(unsigned)((n4 + 255) / 256), 256, 0, stream>>>(out_w, outw16, n4);
    }
    // 1) in_proj: x_inner = x @ in_w^T   (M=8192, N=4096, K=1024)
    {
        dim3 grid(2 * DI / 64, MROWS / 128);
        ssm_gemm_f16_tn<<<grid, 256, 0, stream>>>(x16, inw16, x_inner,
                                                  MROWS, 2 * DI, DM);
    }
    // 2) depthwise conv + silu
    {
        const size_t n = (size_t)MROWS * DI;
        ssm_conv_silu<<<(unsigned)((n + 255) / 256), 256, 0, stream>>>(
            x_inner, conv_w, conv_b, x_conv);
    }
    // 3) delta/B/C projections
    ssm_proj_dbc<<<MROWS, 64, 0, stream>>>(x_conv, delta_w, delta_b, B_w, C_w, dbc);
    // 4) row mean
    ssm_row_mean<<<MROWS, 256, 0, stream>>>(x_conv, xavg);
    // 5) sequential scan (one wave32 per batch)
    ssm_scan<<<BB, 32, 0, stream>>>(dbc, xavg, A_log, yssm);
    // 6) gate + skip -> f16
    {
        const size_t n = (size_t)MROWS * DI;
        ssm_gate_skip<<<(unsigned)((n + 255) / 256), 256, 0, stream>>>(
            x_inner, x_conv, yssm, Dp, yskip16);
    }
    // 7) out_proj: out = y_skip @ out_w^T  (M=8192, N=1024, K=2048)
    {
        dim3 grid(DM / 64, MROWS / 128);
        ssm_gemm_f16_tn<<<grid, 256, 0, stream>>>(yskip16, outw16, out,
                                                  MROWS, DM, DI);
    }
}
